// MemoryBank_35270271435273
// MI455X (gfx1250) — matrix operand order repaired and checked by reading, not compiled, not run on hardware
//
#include <hip/hip_runtime.h>
#include <hip/hip_bf16.h>

typedef __attribute__((ext_vector_type(16))) _Float16 v16h;
typedef __attribute__((ext_vector_type(8)))  _Float16 v8h;
typedef __attribute__((ext_vector_type(8)))  float    v8f;
typedef unsigned int v4u __attribute__((ext_vector_type(4)));
typedef int          v8i __attribute__((ext_vector_type(8)));
typedef int          v4i __attribute__((ext_vector_type(4)));

#define D_DIM 512
#define S_DIM 2048
#define BM    32          // queries per workgroup
#define BN    64          // keys per tile
#define NT    (S_DIM / BN)
#define LDA   520         // padded row stride (halfs) for Qs / Ks
#define LDV   72          // padded row stride (halfs) for Vs / Pbuf
#define LDSB  68          // padded row stride (floats) for score buffer
#define EPSV  1e-12f

// LDS byte offsets (dynamic LDS starts at 0; no static __shared__ in main kernel)
#define QS_OFF 0
#define KS_OFF (BM * LDA * 2)                 // 33280
#define VS_OFF (KS_OFF + BN * LDA * 2)        // 99840
#define PB_OFF (VS_OFF + D_DIM * LDV * 2)     // 173568
#define SB_OFF (PB_OFF + BM * LDV * 2)        // 178176
#define SMEM_BYTES (SB_OFF + BM * LDSB * 4 + 3 * BM * 4)   // 187264

#if defined(__has_builtin)
#  if __has_builtin(__builtin_amdgcn_tensor_load_to_lds) && \
      __has_builtin(__builtin_amdgcn_s_wait_tensorcnt)
#    define HAVE_TDM 1
#  endif
#endif
#ifndef HAVE_TDM
#  define HAVE_TDM 0
#endif

// ---------------------------------------------------------------------------
// Fragment gathers from LDS following CDNA5 WMMA VGPR layouts (ISA 7.12.2)
// ---------------------------------------------------------------------------
__device__ __forceinline__ v16h load_A_frag(const _Float16* base, int row0, int k0, int ld) {
    const int lane = threadIdx.x & 31;
    const int m    = lane & 15;
    const int koff = (lane < 16) ? 0 : 8;
    const _Float16* p = base + (row0 + m) * ld + k0 + koff;
    v8h lo = *(const v8h*)(p);
    v8h hi = *(const v8h*)(p + 16);
    v16h r;
#pragma unroll
    for (int i = 0; i < 8; ++i) { r[i] = lo[i]; r[8 + i] = hi[i]; }
    return r;
}

__device__ __forceinline__ v16h load_B_frag(const _Float16* base, int n0, int k0, int ld) {
    const int lane = threadIdx.x & 31;
    const int n    = lane & 15;
    const int koff = (lane < 16) ? 0 : 8;
    const _Float16* p = base + (n0 + n) * ld + k0 + koff;
    v8h lo = *(const v8h*)(p);
    v8h hi = *(const v8h*)(p + 16);
    v16h r;
#pragma unroll
    for (int i = 0; i < 8; ++i) { r[i] = lo[i]; r[8 + i] = hi[i]; }
    return r;
}

__device__ __forceinline__ v8f zero8() {
    v8f z;
#pragma unroll
    for (int i = 0; i < 8; ++i) z[i] = 0.f;
    return z;
}

#if HAVE_TDM
// ---------------------------------------------------------------------------
// Tensor Data Mover: 2-D tile load, global -> LDS, with LDS row padding.
// D# layout per CDNA5 ISA ch.8 (group0 128b, group1 256b; groups 2/3 zero).
// pad_interval code c: pad after 2^c * 2 DWORDs; pad_amount code a: (a+1) DWORDs.
// 6-arg builtin form (this toolchain): (v4u, v8i, v4i, v4i, v8i, i32 cpol).
// ---------------------------------------------------------------------------
__device__ __forceinline__ void tdm_load_2d(const void* gaddr, unsigned lds_off,
                                            unsigned tensor_d0, unsigned tensor_d1,
                                            unsigned stride0,
                                            unsigned tile_d0, unsigned tile_d1,
                                            unsigned pad_interval, unsigned pad_amount) {
    const unsigned long long ga = (unsigned long long)gaddr;
    v4u g0;
    g0.x = 1u;                                            // count=1, user descriptor
    g0.y = lds_off;                                       // lds_addr (bytes)
    g0.z = (unsigned)(ga & 0xFFFFFFFFu);                  // global_addr[31:0]
    g0.w = (unsigned)((ga >> 32) & 0x01FFFFFFu) | (2u << 30);  // addr[56:32] | type=2
    v8i g1;
    g1[0] = (int)((1u << 16)                              // data_size = 2 bytes
                 | (1u << 20)                             // pad_enable
                 | (pad_interval << 22) | (pad_amount << 25));
    g1[1] = (int)((tensor_d0 & 0xFFFFu) << 16);           // tensor_dim0[15:0]
    g1[2] = (int)((tensor_d0 >> 16) | ((tensor_d1 & 0xFFFFu) << 16));
    g1[3] = (int)((tensor_d1 >> 16) | (tile_d0 << 16));   // | tile_dim0
    g1[4] = (int)tile_d1;                                 // tile_dim1 (tile_dim2 = 0)
    g1[5] = (int)stride0;                                 // tensor_dim0_stride[31:0]
    g1[6] = 0;
    g1[7] = 0;
    const v4i z4 = {0, 0, 0, 0};
    const v8i z8 = {0, 0, 0, 0, 0, 0, 0, 0};
    __builtin_amdgcn_tensor_load_to_lds(g0, g1, z4, z4, z8, 0);
}
#endif

// ---------------------------------------------------------------------------
// Pre-pass 1: L2-normalize keys, emit f16  Kh[S][D]
// ---------------------------------------------------------------------------
__global__ void __launch_bounds__(256)
normalize_keys_kernel(const float* __restrict__ keys, _Float16* __restrict__ Kh) {
    __shared__ float red[8];
    const int s   = blockIdx.x;
    const int tid = threadIdx.x;
    const float* kp = keys + (size_t)s * D_DIM;
    float v0 = kp[tid];
    float v1 = kp[tid + 256];
    float ss = v0 * v0 + v1 * v1;
#pragma unroll
    for (int off = 16; off > 0; off >>= 1) ss += __shfl_xor(ss, off, 32);
    if ((tid & 31) == 0) red[tid >> 5] = ss;
    __syncthreads();
    float tot = 0.f;
#pragma unroll
    for (int i = 0; i < 8; ++i) tot += red[i];
    const float inv = 1.f / fmaxf(sqrtf(tot), EPSV);
    Kh[(size_t)s * D_DIM + tid]       = (_Float16)(v0 * inv);
    Kh[(size_t)s * D_DIM + tid + 256] = (_Float16)(v1 * inv);
}

// ---------------------------------------------------------------------------
// Pre-pass 2: transpose values -> f16  Vt[D][S]  (key dim contiguous)
// ---------------------------------------------------------------------------
__global__ void __launch_bounds__(256)
transpose_v_kernel(const float* __restrict__ values, _Float16* __restrict__ Vt) {
    __shared__ _Float16 tile[64][72];
    const int s0  = blockIdx.x * 64;
    const int d0  = blockIdx.y * 64;
    const int tid = threadIdx.x;
    const int tr  = tid >> 2;
    const int tc  = (tid & 3) * 16;
#pragma unroll
    for (int j = 0; j < 16; ++j)
        tile[tr][tc + j] = (_Float16)values[(size_t)(s0 + tr) * D_DIM + d0 + tc + j];
    __syncthreads();
#pragma unroll
    for (int j = 0; j < 16; ++j)
        Vt[(size_t)(d0 + tr) * S_DIM + s0 + tc + j] = tile[tc + j][tr];
}

// ---------------------------------------------------------------------------
// Main fused cosine-attention kernel (flash style, online softmax).
// 256 threads = 8 waves. TDM double-issues next K/V tiles asynchronously.
// ---------------------------------------------------------------------------
__global__ void __launch_bounds__(256)
cosine_attn_kernel(const float* __restrict__ q,
                   const _Float16* __restrict__ Kh,
                   const _Float16* __restrict__ Vt,
                   float* __restrict__ out) {
    extern __shared__ __align__(16) char smem_raw[];
    _Float16* Qs = (_Float16*)(smem_raw + QS_OFF);   // BM x LDA
    _Float16* Ks = (_Float16*)(smem_raw + KS_OFF);   // BN x LDA
    _Float16* Vs = (_Float16*)(smem_raw + VS_OFF);   // D  x LDV  (Vs[d][k])
    _Float16* Pb = (_Float16*)(smem_raw + PB_OFF);   // BM x LDV
    float*    Sb = (float*)(smem_raw + SB_OFF);      // BM x LDSB
    float*  rowm = Sb + BM * LDSB;
    float*  rowl = rowm + BM;
    float*  corr = rowl + BM;

    const int tid   = threadIdx.x;
    const int wave  = tid >> 5;
    const int lane  = tid & 31;
    const int qbase = blockIdx.x * BM;

#if HAVE_TDM
    // Kick off tile 0 DMA before doing any other work: overlaps Q staging.
    if (wave == 0) {
        tdm_load_2d(Kh, KS_OFF, D_DIM, S_DIM, D_DIM, D_DIM, BN, 7u, 3u);
        tdm_load_2d(Vt, VS_OFF, S_DIM, D_DIM, S_DIM, BN, D_DIM, 4u, 3u);
    }
#endif

    // ---- stage Q block: normalize rows on the fly, store f16 to LDS ----
#pragma unroll
    for (int r4 = 0; r4 < 4; ++r4) {
        const int row = wave * 4 + r4;
        const float* qp = q + (size_t)(qbase + row) * D_DIM;
        float fr[16];
        float ss = 0.f;
#pragma unroll
        for (int j = 0; j < 16; ++j) {
            float v = qp[lane + 32 * j];
            fr[j] = v;
            ss += v * v;
        }
#pragma unroll
        for (int off = 16; off > 0; off >>= 1) ss += __shfl_xor(ss, off, 32);
        const float inv = 1.f / fmaxf(sqrtf(ss), EPSV);
#pragma unroll
        for (int j = 0; j < 16; ++j)
            Qs[row * LDA + lane + 32 * j] = (_Float16)(fr[j] * inv);
    }
    if (tid < BM) { rowm[tid] = -3.0e38f; rowl[tid] = 0.f; }

    v8f o[8];
#pragma unroll
    for (int t = 0; t < 8; ++t) o[t] = zero8();

    const int sm = wave & 1;   // score M-group
    const int sn = wave >> 1;  // score key-group (0..3)
    const int wm = wave & 1;   // output M-group
    const int dn = wave >> 1;  // output d-group (0..3, 128 wide)

    __syncthreads();

    for (int kt = 0; kt < NT; ++kt) {
        const int kbase = kt * BN;

#if HAVE_TDM
        // Drain outstanding tile DMA (K/V for this iteration), then publish.
        if (wave == 0) __builtin_amdgcn_s_wait_tensorcnt(0);
        __syncthreads();
#else
        if (kt + 1 < NT) {
            const _Float16* nk = Kh + (size_t)(kbase + BN) * D_DIM;
            __builtin_prefetch((const char*)nk + tid * 256, 0, 1);
        }
        for (int i = tid; i < BN * D_DIM / 8; i += 256) {
            const int idx = i * 8;
            const int key = idx >> 9;
            const int d   = idx & 511;
            *(v8h*)(Ks + key * LDA + d) =
                *(const v8h*)(Kh + (size_t)(kbase + key) * D_DIM + d);
        }
        for (int i = tid; i < D_DIM * BN / 8; i += 256) {
            const int idx = i * 8;
            const int d = idx >> 6;
            const int k = idx & 63;
            *(v8h*)(Vs + d * LDV + k) =
                *(const v8h*)(Vt + (size_t)d * S_DIM + kbase + k);
        }
        __syncthreads();
#endif

        // ---- scores: one 16x16 tile per wave, accumulate over D ----
        v8f sc = zero8();
#pragma unroll
        for (int kk = 0; kk < D_DIM; kk += 32) {
            v16h a = load_A_frag(Qs, 16 * sm, kk, LDA);
            v16h b = load_B_frag(Ks, 16 * sn, kk, LDA);
            sc = __builtin_amdgcn_wmma_f32_16x16x32_f16(false, a, false, b,
                                                        (short)0, sc, false, false);
        }
        {
            const int r0 = 16 * sm + ((lane < 16) ? 0 : 8);
            const int c  = 16 * sn + (lane & 15);
#pragma unroll
            for (int r = 0; r < 8; ++r) Sb[(r0 + r) * LDSB + c] = sc[r];
        }
        __syncthreads();   // all waves done reading Ks, Sb complete

#if HAVE_TDM
        // Ks is dead now: start async DMA of next K tile (overlaps softmax+PV).
        if (wave == 0 && kt + 1 < NT)
            tdm_load_2d(Kh + (size_t)(kbase + BN) * D_DIM, KS_OFF,
                        D_DIM, S_DIM, D_DIM, D_DIM, BN, 7u, 3u);
#endif

        // ---- online softmax over this key tile (wave 0, one row per lane) ----
        if (wave == 0) {
            const int row = lane;
            const float mo = rowm[row];
            const float lo = rowl[row];
            float mx = mo;
#pragma unroll
            for (int j = 0; j < BN; ++j) mx = fmaxf(mx, Sb[row * LDSB + j]);
            const float c = __expf(mo - mx);
            float s = 0.f;
#pragma unroll
            for (int j = 0; j < BN; ++j) {
                const float p = __expf(Sb[row * LDSB + j] - mx);
                s += p;
                Pb[row * LDV + j] = (_Float16)p;
            }
            rowm[row] = mx;
            rowl[row] = lo * c + s;
            corr[row] = c;
        }
        __syncthreads();

        // ---- rescale accumulators, then O += P @ V ----
        {
            float cf[8];
            const int r0 = 16 * wm + ((lane < 16) ? 0 : 8);
#pragma unroll
            for (int r = 0; r < 8; ++r) cf[r] = corr[r0 + r];

            v16h a0 = load_A_frag(Pb, 16 * wm, 0, LDV);
            v16h a1 = load_A_frag(Pb, 16 * wm, 32, LDV);
#pragma unroll
            for (int t = 0; t < 8; ++t) {
#pragma unroll
                for (int r = 0; r < 8; ++r) o[t][r] *= cf[r];
                const int nb = 128 * dn + 16 * t;
                v16h b0 = load_B_frag(Vs, nb, 0, LDV);
                v16h b1 = load_B_frag(Vs, nb, 32, LDV);
                o[t] = __builtin_amdgcn_wmma_f32_16x16x32_f16(false, a0, false, b0,
                                                              (short)0, o[t], false, false);
                o[t] = __builtin_amdgcn_wmma_f32_16x16x32_f16(false, a1, false, b1,
                                                              (short)0, o[t], false, false);
            }
        }
        __syncthreads();   // Vs / Pb dead beyond this point

#if HAVE_TDM
        // Start async DMA of next V tile.
        if (wave == 0 && kt + 1 < NT)
            tdm_load_2d(Vt + (kbase + BN), VS_OFF,
                        S_DIM, D_DIM, S_DIM, BN, D_DIM, 4u, 3u);
#endif
    }

    // ---- epilogue: divide by softmax denominator, store fp32 ----
    {
        const int r0 = 16 * wm + ((lane < 16) ? 0 : 8);
        float li[8];
#pragma unroll
        for (int r = 0; r < 8; ++r) li[r] = 1.f / rowl[r0 + r];
#pragma unroll
        for (int t = 0; t < 8; ++t) {
            const int col = 128 * dn + 16 * t + (lane & 15);
#pragma unroll
            for (int r = 0; r < 8; ++r)
                out[(size_t)(qbase + r0 + r) * D_DIM + col] = o[t][r] * li[r];
        }
    }
}

// ---------------------------------------------------------------------------
extern "C" void kernel_launch(void* const* d_in, const int* in_sizes, int n_in,
                              void* d_out, int out_size, void* d_ws, size_t ws_size,
                              hipStream_t stream) {
    const float* q      = (const float*)d_in[0];
    const float* keys   = (const float*)d_in[1];
    const float* values = (const float*)d_in[2];
    float* out          = (float*)d_out;

    const int N = in_sizes[0] / D_DIM;   // 65536

    _Float16* Kh = (_Float16*)d_ws;                       // S*D halfs (2 MB)
    _Float16* Vt = Kh + (size_t)S_DIM * D_DIM;            // D*S halfs (2 MB)

    normalize_keys_kernel<<<S_DIM, 256, 0, stream>>>(keys, Kh);
    dim3 tg(S_DIM / 64, D_DIM / 64);
    transpose_v_kernel<<<tg, 256, 0, stream>>>(values, Vt);

    (void)hipFuncSetAttribute((const void*)cosine_attn_kernel,
                              hipFuncAttributeMaxDynamicSharedMemorySize,
                              (int)SMEM_BYTES);

    cosine_attn_kernel<<<N / BM, 256, SMEM_BYTES, stream>>>(q, Kh, Vt, out);
}
